// TriangleAttention_60017872994854
// MI455X (gfx1250) — compile-verified, matrix operand order checked by reading, not run on hardware
//
#include <hip/hip_runtime.h>
#include <stdint.h>

#define L 256
#define D 128
#define H 4
#define DH 32
#define NPOS (L * L)   // 65536
#define HD (H * DH)    // 128

typedef __attribute__((ext_vector_type(16))) __bf16 v16bf;
typedef __attribute__((ext_vector_type(8)))  float  v8f;

union FragBF { v16bf v; uint4 u[2]; };

__device__ __forceinline__ unsigned short f2bf(float f) {
  unsigned int u = __float_as_uint(f);
  u += 0x7FFFu + ((u >> 16) & 1u);   // round-to-nearest-even
  return (unsigned short)(u >> 16);
}
__device__ __forceinline__ float bf2f(unsigned short h) {
  return __uint_as_float(((unsigned int)h) << 16);
}

// CDNA5 async global->LDS copy (ASYNCcnt-tracked, no VGPR round trip).
__device__ __forceinline__ void async_g2l_b128(unsigned lds_off, uint64_t gaddr) {
  asm volatile("global_load_async_to_lds_b128 %0, %1, off"
               :: "v"(lds_off), "v"(gaddr) : "memory");
}
__device__ __forceinline__ void wait_async0() {
  asm volatile("s_wait_asynccnt 0x0" ::: "memory");
}

// ---------------------------------------------------------------- K0: weights
// Pack Wq|Wk|Wv|Wg into Wt[n][k] (512x128 bf16, transposed) and Wo into
// WoT[n][k] (128x128 bf16) so WMMA B-fragments are two contiguous b128 loads.
__global__ __launch_bounds__(256) void prep_weights(
    const float* __restrict__ Wq, const float* __restrict__ Wk,
    const float* __restrict__ Wv, const float* __restrict__ Wg,
    const float* __restrict__ Wo,
    unsigned short* __restrict__ Wt, unsigned short* __restrict__ WoT) {
  int t = blockIdx.x * 256 + threadIdx.x;
  if (t < 512 * 128) {
    int n = t >> 7, k = t & 127;
    float w;
    if (n < 128)      w = Wq[k * 128 + n];
    else if (n < 256) w = Wk[k * 128 + (n - 128)];
    else if (n < 384) w = Wv[k * 128 + (n - 256)];
    else              w = Wg[k * 128 + (n - 384)];
    Wt[t] = f2bf(w);
  } else if (t < 512 * 128 + 128 * 128) {
    int t2 = t - 512 * 128;
    int n = t2 >> 7, k = t2 & 127;
    WoT[t2] = f2bf(Wo[k * 128 + n]);
  }
}

// ---------------------------------------------------------------- K1: LayerNorm
__global__ __launch_bounds__(256) void ln_kernel(
    const float* __restrict__ pair, const float* __restrict__ ln_w,
    const float* __restrict__ ln_b, unsigned short* __restrict__ xb) {
  int wid = threadIdx.x >> 5, lane = threadIdx.x & 31;
  size_t row = (size_t)blockIdx.x * 8 + wid;
  float4 x = *(const float4*)(pair + row * D + lane * 4);
  float s = x.x + x.y + x.z + x.w;
  for (int o = 1; o < 32; o <<= 1) s += __shfl_xor(s, o, 32);
  float mu = s * (1.f / D);
  float d0 = x.x - mu, d1 = x.y - mu, d2 = x.z - mu, d3 = x.w - mu;
  float vv = d0 * d0 + d1 * d1 + d2 * d2 + d3 * d3;
  for (int o = 1; o < 32; o <<= 1) vv += __shfl_xor(vv, o, 32);
  float rstd = rsqrtf(vv * (1.f / D) + 1e-5f);
  float4 w = *(const float4*)(ln_w + lane * 4);
  float4 b = *(const float4*)(ln_b + lane * 4);
  unsigned int lo = (unsigned int)f2bf(d0 * rstd * w.x + b.x) |
                    ((unsigned int)f2bf(d1 * rstd * w.y + b.y) << 16);
  unsigned int hi = (unsigned int)f2bf(d2 * rstd * w.z + b.z) |
                    ((unsigned int)f2bf(d3 * rstd * w.w + b.w) << 16);
  uint2 packed; packed.x = lo; packed.y = hi;
  *(uint2*)(xb + row * D + lane * 4) = packed;
}

// ---------------------------------------------------------------- K2: Q/K/V/G GEMM
__global__ __launch_bounds__(256) void gemm_qkvg(
    const unsigned short* __restrict__ xb, const unsigned short* __restrict__ Wt,
    const float* __restrict__ bg,
    unsigned short* __restrict__ qb, unsigned short* __restrict__ kbuf,
    unsigned short* __restrict__ vt, unsigned short* __restrict__ gb) {
  int wid = threadIdx.x >> 5, lane = threadIdx.x & 31;
  int mt = blockIdx.x;
  int nt = blockIdx.y * 8 + wid;        // 0..31
  int mr = lane & 15;
  int kb8 = (lane >> 4) << 3;
  size_t arow = ((size_t)mt * 16 + mr) * (size_t)D;
  size_t brow = ((size_t)nt * 16 + mr) * (size_t)D;
  v8f acc = {};
#pragma unroll
  for (int kk = 0; kk < D; kk += 32) {
    FragBF a, b;
    a.u[0] = *(const uint4*)(xb + arow + kk + kb8);
    a.u[1] = *(const uint4*)(xb + arow + kk + 16 + kb8);
    b.u[0] = *(const uint4*)(Wt + brow + kk + kb8);
    b.u[1] = *(const uint4*)(Wt + brow + kk + 16 + kb8);
    acc = __builtin_amdgcn_wmma_f32_16x16x32_bf16(false, a.v, false, b.v,
                                                  (short)0, acc, false, false);
  }
  const float qscale = 0.17677669529663687f;  // DH^-0.5
  // seg is wave-uniform: force into SGPR so routing is a scalar branch
  int seg = __builtin_amdgcn_readfirstlane(nt >> 3);
  int nin = (nt & 7) * 16 + (lane & 15);  // column within 128-wide segment
#pragma unroll
  for (int r = 0; r < 8; ++r) {
    int mm = r + ((lane >> 4) << 3);
    size_t pos = (size_t)mt * 16 + mm;
    float vacc = acc[r];
    if (seg == 0) {
      qb[pos * HD + nin] = f2bf(vacc * qscale);
    } else if (seg == 1) {
      kbuf[pos * HD + nin] = f2bf(vacc);
    } else if (seg == 2) {
      int hh = nin >> 5, dd = nin & 31;
      int ii = (int)(pos >> 8), kp = (int)(pos & 255);
      vt[(((size_t)ii * H + hh) * DH + dd) * L + kp] = f2bf(vacc);
    } else {
      float g = vacc + bg[nin];
      gb[pos * HD + nin] = f2bf(1.f / (1.f + __expf(-g)));
    }
  }
}

// ---------------------------------------------------------------- K2b: pair bias
// bias stored [h][j][k] so attention loads are lane-coalesced.
__global__ __launch_bounds__(256) void bias_kernel(
    const unsigned short* __restrict__ xb, const float* __restrict__ Wb,
    float* __restrict__ biasb) {
  int t = blockIdx.x * 256 + threadIdx.x;  // < NPOS*H
  int pos = t >> 2, h = t & 3;
  const unsigned short* xr = xb + (size_t)pos * D;
  float s = 0.f;
#pragma unroll 8
  for (int k = 0; k < D; ++k) s += bf2f(xr[k]) * Wb[k * H + h];
  biasb[(size_t)h * NPOS + pos] = s;
}

// ---------------------------------------------------------------- K3: attention
// Flash-style per (i, h); all 8 waves share (i,h) so K/V chunks are staged
// into LDS once per block via async global->LDS B128 copies, double-buffered
// (issue chunk c+1, compute chunk c). P goes through wave-local LDS to turn
// the WMMA C-layout into the A-layout.
__global__ __launch_bounds__(256) void attn_kernel(
    const unsigned short* __restrict__ qb, const unsigned short* __restrict__ kbuf,
    const unsigned short* __restrict__ vt, const unsigned short* __restrict__ gb,
    const float* __restrict__ biasb, unsigned short* __restrict__ ao) {
  __shared__ unsigned short ldsK[2 * 32 * 32];   // double-buffered K chunk
  __shared__ unsigned short ldsV[2 * 32 * 32];   // double-buffered V chunk
  __shared__ unsigned short ldsP[8 * 16 * 32];   // per-wave P staging

  int tid = threadIdx.x;
  int wid = tid >> 5, lane = tid & 31;
  int i = blockIdx.z, h = blockIdx.y;
  int j0 = blockIdx.x * 128 + wid * 16;
  int nn = lane & 15;
  int kb8 = (lane >> 4) << 3;
  size_t rowbase = (size_t)i * L;
  size_t vbase = ((size_t)(i * H + h) * DH) * L;

  unsigned ldsKoff = (unsigned)(uintptr_t)(&ldsK[0]);
  unsigned ldsVoff = (unsigned)(uintptr_t)(&ldsV[0]);

  // cooperative async issue of one 32-wide k chunk into buffer `buf`
  auto issue_chunk = [&](int k0, int buf) {
    if (tid < 128) {               // waves 0-3: K rows (32 rows x 32 dims)
      int r = tid >> 2, p = tid & 3;
      async_g2l_b128(ldsKoff + buf * 2048 + r * 64 + p * 16,
                     (uint64_t)(kbuf + (rowbase + k0 + r) * HD + h * DH + p * 8));
    } else {                       // waves 4-7: V^T rows (32 d x 32 k)
      int t2 = tid - 128;
      int d = t2 >> 2, p = t2 & 3;
      async_g2l_b128(ldsVoff + buf * 2048 + d * 64 + p * 16,
                     (uint64_t)(vt + vbase + (size_t)d * L + k0 + p * 8));
    }
  };

  FragBF aq;  // Q rows j0..j0+15, 32 dims (scale pre-applied in K2)
  size_t qrow = (rowbase + j0 + nn) * HD + h * DH;
  aq.u[0] = *(const uint4*)(qb + qrow + kb8);
  aq.u[1] = *(const uint4*)(qb + qrow + 16 + kb8);

  float rmax[8], rsum[8];
  v8f o0 = {}, o1 = {};
#pragma unroll
  for (int r = 0; r < 8; ++r) { rmax[r] = -1e30f; rsum[r] = 0.f; }

  unsigned short* myP = ldsP + wid * (16 * 32);
  const float* biash = biasb + (size_t)h * NPOS;

  issue_chunk(0, 0);

  for (int c = 0; c < L / 32; ++c) {
    int k0 = c * 32;
    int buf = c & 1;
    wait_async0();
    __syncthreads();                 // chunk c resident; prev buffer free
    if (c + 1 < L / 32) issue_chunk(k0 + 32, buf ^ 1);

    const unsigned short* cK = ldsK + buf * 1024;
    const unsigned short* cV = ldsV + buf * 1024;

    // --- S = Q.K^T for k columns [k0,k0+32) ------------------------------
    FragBF bk0, bk1;
    bk0.u[0] = *(const uint4*)(cK + nn * 32 + kb8);
    bk0.u[1] = *(const uint4*)(cK + nn * 32 + 16 + kb8);
    bk1.u[0] = *(const uint4*)(cK + (16 + nn) * 32 + kb8);
    bk1.u[1] = *(const uint4*)(cK + (16 + nn) * 32 + 16 + kb8);
    v8f z = {};
    v8f s0 = __builtin_amdgcn_wmma_f32_16x16x32_bf16(false, aq.v, false, bk0.v,
                                                     (short)0, z, false, false);
    v8f s1 = __builtin_amdgcn_wmma_f32_16x16x32_bf16(false, aq.v, false, bk1.v,
                                                     (short)0, z, false, false);
    // --- add pair bias (layout [h][j][k]) --------------------------------
#pragma unroll
    for (int r = 0; r < 8; ++r) {
      int mm = r + ((lane >> 4) << 3);
      const float* bp = biash + (size_t)(j0 + mm) * L + k0 + nn;
      s0[r] += bp[0];
      s1[r] += bp[16];
    }
    // --- online softmax update -------------------------------------------
#pragma unroll
    for (int r = 0; r < 8; ++r) {
      float cm = fmaxf(s0[r], s1[r]);
      for (int o = 1; o < 16; o <<= 1) cm = fmaxf(cm, __shfl_xor(cm, o, 32));
      float nm = fmaxf(rmax[r], cm);
      float corr = __expf(rmax[r] - nm);
      rmax[r] = nm;
      float p0 = __expf(s0[r] - nm);
      float p1 = __expf(s1[r] - nm);
      float ps = p0 + p1;
      for (int o = 1; o < 16; o <<= 1) ps += __shfl_xor(ps, o, 32);
      rsum[r] = rsum[r] * corr + ps;
      o0[r] *= corr;
      o1[r] *= corr;
      int mm = r + ((lane >> 4) << 3);
      myP[mm * 32 + nn]      = f2bf(p0);   // C-layout -> LDS
      myP[mm * 32 + 16 + nn] = f2bf(p1);
    }
    // --- O += P.V (reload P in A-layout; V chunk from LDS) ---------------
    FragBF ap;
    ap.u[0] = *(const uint4*)(myP + nn * 32 + kb8);
    ap.u[1] = *(const uint4*)(myP + nn * 32 + 16 + kb8);
    FragBF bv0, bv1;
    bv0.u[0] = *(const uint4*)(cV + nn * 32 + kb8);
    bv0.u[1] = *(const uint4*)(cV + nn * 32 + 16 + kb8);
    bv1.u[0] = *(const uint4*)(cV + (16 + nn) * 32 + kb8);
    bv1.u[1] = *(const uint4*)(cV + (16 + nn) * 32 + 16 + kb8);
    o0 = __builtin_amdgcn_wmma_f32_16x16x32_bf16(false, ap.v, false, bv0.v,
                                                 (short)0, o0, false, false);
    o1 = __builtin_amdgcn_wmma_f32_16x16x32_bf16(false, ap.v, false, bv1.v,
                                                 (short)0, o1, false, false);
  }
  // --- normalize, gate, store bf16 for final GEMM ------------------------
#pragma unroll
  for (int r = 0; r < 8; ++r) {
    int mm = r + ((lane >> 4) << 3);
    size_t pos = rowbase + j0 + mm;
    float inv = 1.f / rsum[r];
    int c0 = h * DH + nn, c1 = c0 + 16;
    float g0 = bf2f(gb[pos * HD + c0]);
    float g1 = bf2f(gb[pos * HD + c1]);
    ao[pos * HD + c0] = f2bf(o0[r] * inv * g0);
    ao[pos * HD + c1] = f2bf(o1[r] * inv * g1);
  }
}

// ---------------------------------------------------------------- K4: out GEMM
__global__ __launch_bounds__(256) void gemm_out(
    const unsigned short* __restrict__ ao, const unsigned short* __restrict__ WoT,
    const float* __restrict__ bo, float* __restrict__ out) {
  int wid = threadIdx.x >> 5, lane = threadIdx.x & 31;
  int mt = blockIdx.x, nt = wid;  // 8 n-tiles = 128 cols
  int mr = lane & 15, kb8 = (lane >> 4) << 3;
  size_t arow = ((size_t)mt * 16 + mr) * (size_t)HD;
  size_t brow = ((size_t)nt * 16 + mr) * (size_t)HD;
  v8f acc = {};
#pragma unroll
  for (int kk = 0; kk < HD; kk += 32) {
    FragBF a, b;
    a.u[0] = *(const uint4*)(ao + arow + kk + kb8);
    a.u[1] = *(const uint4*)(ao + arow + kk + 16 + kb8);
    b.u[0] = *(const uint4*)(WoT + brow + kk + kb8);
    b.u[1] = *(const uint4*)(WoT + brow + kk + 16 + kb8);
    acc = __builtin_amdgcn_wmma_f32_16x16x32_bf16(false, a.v, false, b.v,
                                                  (short)0, acc, false, false);
  }
#pragma unroll
  for (int r = 0; r < 8; ++r) {
    int mm = r + ((lane >> 4) << 3);
    size_t pos = (size_t)mt * 16 + mm;
    int n = nt * 16 + (lane & 15);
    out[pos * D + n] = acc[r] + bo[n];
  }
}

// ---------------------------------------------------------------- launch
extern "C" void kernel_launch(void* const* d_in, const int* in_sizes, int n_in,
                              void* d_out, int out_size, void* d_ws, size_t ws_size,
                              hipStream_t stream) {
  (void)in_sizes; (void)n_in; (void)out_size; (void)ws_size;
  const float* pair = (const float*)d_in[0];
  const float* ln_w = (const float*)d_in[1];
  const float* ln_b = (const float*)d_in[2];
  const float* Wq   = (const float*)d_in[3];
  const float* Wk   = (const float*)d_in[4];
  const float* Wv   = (const float*)d_in[5];
  const float* Wb   = (const float*)d_in[6];
  const float* Wg   = (const float*)d_in[7];
  const float* bg   = (const float*)d_in[8];
  const float* Wo   = (const float*)d_in[9];
  const float* bo   = (const float*)d_in[10];
  float* out = (float*)d_out;

  char* ws = (char*)d_ws;
  size_t off = 0;
  auto alloc = [&](size_t bytes) {
    char* p = ws + off;
    off += (bytes + 255) & ~(size_t)255;
    return p;
  };
  unsigned short* xb    = (unsigned short*)alloc((size_t)NPOS * D * 2);
  unsigned short* qb    = (unsigned short*)alloc((size_t)NPOS * HD * 2);
  unsigned short* kbuf  = (unsigned short*)alloc((size_t)NPOS * HD * 2);
  unsigned short* vt    = (unsigned short*)alloc((size_t)NPOS * HD * 2);
  unsigned short* gb    = (unsigned short*)alloc((size_t)NPOS * HD * 2);
  unsigned short* ao    = (unsigned short*)alloc((size_t)NPOS * HD * 2);
  float*          biasb = (float*)alloc((size_t)NPOS * H * 4);
  unsigned short* Wt    = (unsigned short*)alloc((size_t)512 * 128 * 2);
  unsigned short* WoT   = (unsigned short*)alloc((size_t)128 * 128 * 2);

  prep_weights<<<(512 * 128 + 128 * 128 + 255) / 256, 256, 0, stream>>>(
      Wq, Wk, Wv, Wg, Wo, Wt, WoT);
  ln_kernel<<<NPOS / 8, 256, 0, stream>>>(pair, ln_w, ln_b, xb);
  gemm_qkvg<<<dim3(NPOS / 16, 4), 256, 0, stream>>>(xb, Wt, bg, qb, kbuf, vt, gb);
  bias_kernel<<<(NPOS * H) / 256, 256, 0, stream>>>(xb, Wb, biasb);
  attn_kernel<<<dim3(2, H, L), 256, 0, stream>>>(qb, kbuf, vt, gb, biasb, ao);
  gemm_out<<<NPOS / 16, 256, 0, stream>>>(ao, WoT, bo, out);
}
